// SLinOSSMixer_40037685133621
// MI455X (gfx1250) — matrix operand order repaired and checked by reading, not compiled, not run on hardware
//
#include <hip/hip_runtime.h>
#include <hip/hip_bf16.h>

// ---------------- problem constants ----------------
constexpr int T    = 1024;
constexpr int DM   = 768;
constexpr int DI   = 1536;
constexpr int H    = 24;         // DI / DH
constexpr int N    = 64;
constexpr int DH   = 64;
constexpr int DC   = 4;
constexpr int P    = 2 * DI + H; // 3096 (true proj width)
constexpr int Ppad = 3104;       // padded to multiple of 32 for WMMA GEMM
constexpr int BCW  = H * 4 * N;  // 6144
constexpr float DT_MIN = 1e-4f;
constexpr float DT_MAX = 0.1f;
constexpr float EPS    = 1e-5f;

typedef __bf16 bf16;
typedef __attribute__((ext_vector_type(16))) __bf16 v16bf;
typedef __attribute__((ext_vector_type(8)))  float  v8f;

__device__ __forceinline__ float softplusf(float x) {
    return (x > 20.f) ? x : log1pf(__expf(x));
}
__device__ __forceinline__ float siluf(float x) {
    return x / (1.f + __expf(-x));
}

// ---------------- f32 -> bf16 convert (flat) ----------------
__global__ __launch_bounds__(256) void k_cvt_bf16(const float* __restrict__ in,
                                                  bf16* __restrict__ out, int n) {
    int i = blockIdx.x * 256 + threadIdx.x;
    if (i < n) out[i] = (bf16)in[i];
}

// ---------------- f32 [K,No] -> bf16 transposed [NoPad,K], zero pad ----------------
// 32x32 LDS-tiled transpose: coalesced reads and writes.
// grid = (K/32, NoPad/32), 256 threads.
__global__ __launch_bounds__(256) void k_transpose_cvt(const float* __restrict__ in,
                                                       bf16* __restrict__ out,
                                                       int K, int No, int NoPad) {
    __shared__ float tile[32][33];
    const int kt = blockIdx.x, ct = blockIdx.y;
    const int tx = threadIdx.x & 31, ty = threadIdx.x >> 5;   // 8 rows per pass
#pragma unroll
    for (int j = 0; j < 4; ++j) {
        int row = ty + j * 8;              // k within tile
        int col = ct * 32 + tx;            // No index
        tile[row][tx] = (col < No) ? in[(size_t)(kt * 32 + row) * No + col] : 0.f;
    }
    __syncthreads();
#pragma unroll
    for (int j = 0; j < 4; ++j) {
        int row = ty + j * 8;              // No-local (output row)
        out[(size_t)(ct * 32 + row) * K + kt * 32 + tx] = (bf16)tile[tx][row];
    }
}

// ---------------- WMMA bf16 GEMM: out[MxNo] = A[MxK] * Bt[NoxK]^T ----------------
// A row-major [M,K] bf16; Bt is the weight TRANSPOSED to [No,K] bf16 so that
// every fragment (A and B) is two contiguous global_load_b128 per lane.
// One wave computes a 64Mx32N macro-tile: 4 A frags x 2 B frags -> 8 v_wmma
// per 32-wide K-step. M multiple of 64; No, K multiples of 32.
// Fragment layouts per CDNA5 ISA 7.12.2 (wave32):
//   A (16x32 bf16): lane -> M=lane&15; dword g: K-pair = (g&4)*4+(g&3)*2+(lane>>4)*8
//   B (32x16 bf16): lane -> N=lane&15; dword g: same K-pair as A (row of Bt)
//   C/D (16x16 f32): VGPR r -> M = r + (lane>>4)*8, N = lane&15
__global__ __launch_bounds__(256) void k_gemm_bf16(float* __restrict__ out,
                                                   const bf16* __restrict__ A,
                                                   const bf16* __restrict__ Bt,
                                                   int M, int No, int K) {
    const int lane = threadIdx.x & 31;
    const int wave = threadIdx.x >> 5;
    const int nTN  = No >> 5;            // 32-wide column tiles
    const int nTM  = M >> 6;             // 64-high row tiles
    const int wgid = blockIdx.x * 8 + wave;
    if (wgid >= nTM * nTN) return;
    const int tN = wgid % nTN;
    const int tM = wgid / nTN;

    const int hi = lane >> 4;
    const int lm = lane & 15;

    const uint* ap[4];
#pragma unroll
    for (int j = 0; j < 4; ++j)
        ap[j] = (const uint*)(A + (size_t)(tM * 64 + j * 16 + lm) * K);
    const uint* bp[2];
#pragma unroll
    for (int i = 0; i < 2; ++i)
        bp[i] = (const uint*)(Bt + (size_t)(tN * 32 + i * 16 + lm) * K);

    v8f acc[4][2] = {};

    for (int k0 = 0; k0 < K; k0 += 32) {
        const int kd = k0 >> 1;          // dword offset of this K-chunk
        if (k0 + 32 < K) {
            __builtin_prefetch(ap[0] + kd + 16, 0, 1);   // global_prefetch_b8
            __builtin_prefetch(bp[0] + kd + 16, 0, 1);
        }

        union Frag { uint u[8]; v16bf v; };
        Frag a[4], b[2];
#pragma unroll
        for (int j = 0; j < 4; ++j)
#pragma unroll
            for (int g = 0; g < 8; ++g) {
                const int d = ((g & 4) << 1) + (g & 3) + (hi << 2);
                a[j].u[g] = ap[j][kd + d];
            }
#pragma unroll
        for (int i = 0; i < 2; ++i)
#pragma unroll
            for (int g = 0; g < 8; ++g)
                b[i].u[g] = bp[i][kd + ((g & 4) << 1) + (g & 3) + (hi << 2)];

#pragma unroll
        for (int j = 0; j < 4; ++j)
#pragma unroll
            for (int i = 0; i < 2; ++i)
                acc[j][i] = __builtin_amdgcn_wmma_f32_16x16x32_bf16(
                    false, a[j].v, false, b[i].v, (short)0, acc[j][i], false, false);
    }

#pragma unroll
    for (int j = 0; j < 4; ++j) {
        const int rowBase = tM * 64 + j * 16 + hi * 8;
#pragma unroll
        for (int i = 0; i < 2; ++i) {
            const int col = tN * 32 + i * 16 + lm;
#pragma unroll
            for (int r = 0; r < 8; ++r)
                out[(size_t)(rowBase + r) * No + col] = acc[j][i][r];
        }
    }
}

// ---------------- depthwise causal conv (DC taps) + SiLU ----------------
__global__ __launch_bounds__(256) void k_conv_silu(const float* __restrict__ proj,
                                                   const float* __restrict__ conv_w,
                                                   const float* __restrict__ conv_b,
                                                   float* __restrict__ value,
                                                   bf16* __restrict__ valb) {
    int i = blockIdx.x * 256 + threadIdx.x;
    if (i >= T * DI) return;
    int t = i / DI;
    int c = i - t * DI;
    float acc = conv_b[c];
#pragma unroll
    for (int kk = 0; kk < DC; ++kk) {
        int tt = t + kk - (DC - 1);
        if (tt >= 0) acc += proj[(size_t)tt * Ppad + DI + c] * conv_w[c * DC + kk];
    }
    float v = siluf(acc);
    value[i] = v;
    valb[i]  = (bf16)v;
}

// ---------------- RMS-norm rows of bc (in place) * scales ----------------
__global__ __launch_bounds__(256) void k_rmsnorm(float* __restrict__ bc,
                                                 const float* __restrict__ b_scale,
                                                 const float* __restrict__ c_scale) {
    int wave = threadIdx.x >> 5, lane = threadIdx.x & 31;
    int row = blockIdx.x * 8 + wave;           // T*H*4 rows
    if (row >= T * H * 4) return;
    int rem = row % (H * 4);
    int h = rem >> 2, j = rem & 3;

    float* p = bc + (size_t)row * N;
    float v0 = p[lane], v1 = p[lane + 32];
    float ss = v0 * v0 + v1 * v1;
#pragma unroll
    for (int m = 16; m >= 1; m >>= 1) ss += __shfl_xor(ss, m, 32);
    float r = rsqrtf(ss * (1.f / 64.f) + EPS);

    const float* sc = (j < 2) ? (b_scale + (h * 2 + j) * N)
                              : (c_scale + (h * 2 + (j - 2)) * N);
    p[lane]      = v0 * r * sc[lane];
    p[lane + 32] = v1 * r * sc[lane + 32];
}

// ---------------- sequential scan over T, one block per head ----------------
__global__ __launch_bounds__(256) void k_scan(const float* __restrict__ proj,
                                              const float* __restrict__ bc,
                                              const float* __restrict__ value,
                                              const float* __restrict__ A_log,
                                              const float* __restrict__ dt_bias,
                                              float* __restrict__ scan_y) {
    const int h   = blockIdx.x;      // 0..H-1
    const int tid = threadIdx.x;
    const int dh  = tid >> 2;
    const int r   = tid & 3;
    const int nb  = r * 16;

    __shared__ float mf[8][N];   // m0,m1,m2,m3,f0,f1,c0,c1 per n
    __shared__ float uS[DH];

    float s0[16], s1[16];
#pragma unroll
    for (int i = 0; i < 16; ++i) { s0[i] = 0.f; s1[i] = 0.f; }

    float An = 0.f, dtb = dt_bias[h];
    if (tid < N) An = softplusf(A_log[h * N + tid]);

    for (int t = 0; t < T; ++t) {
        if (tid < N) {
            const int n = tid;
            float pval = proj[(size_t)t * Ppad + 2 * DI + h];
            float dt = softplusf(pval + dtb);
            dt = fminf(fmaxf(dt, DT_MIN), DT_MAX);
            float A = An;
            float S = 1.f / (1.f + dt * dt * A);
            const float* row = bc + ((size_t)(t * H + h) * 4) * N;
            float B0 = row[n], B1 = row[N + n], C0 = row[2 * N + n], C1 = row[3 * N + n];
            mf[0][n] = S;
            mf[1][n] = -dt * A * S;
            mf[2][n] = dt * S;
            mf[3][n] = 1.f - dt * dt * A * S;
            mf[4][n] = dt * S * B0;
            mf[5][n] = dt * dt * S * B1;
            mf[6][n] = C0;
            mf[7][n] = C1;
        } else if (tid < N + DH) {
            const int d = tid - N;
            uS[d] = value[(size_t)t * DI + h * DH + d];
        }
        __syncthreads();

        const float u = uS[dh];
        float yacc = 0.f;
#pragma unroll
        for (int i = 0; i < 16; ++i) {
            const int n = nb + i;
            float ns0 = mf[0][n] * s0[i] + mf[1][n] * s1[i] + mf[4][n] * u;
            float ns1 = mf[2][n] * s0[i] + mf[3][n] * s1[i] + mf[5][n] * u;
            s0[i] = ns0; s1[i] = ns1;
            yacc += mf[6][n] * ns0 + mf[7][n] * ns1;
        }
        yacc += __shfl_xor(yacc, 1, 32);
        yacc += __shfl_xor(yacc, 2, 32);
        if (r == 0) scan_y[(size_t)t * DI + h * DH + dh] = yacc;
        __syncthreads();
    }
}

// ---------------- combine: yc = (scan_y + value*skip) * silu(gate) ----------------
__global__ __launch_bounds__(256) void k_combine(const float* __restrict__ proj,
                                                 const float* __restrict__ scan_y,
                                                 const float* __restrict__ value,
                                                 const float* __restrict__ skip,
                                                 bf16* __restrict__ ycb) {
    int i = blockIdx.x * 256 + threadIdx.x;
    if (i >= T * DI) return;
    int t = i / DI;
    int c = i - t * DI;
    float g = proj[(size_t)t * Ppad + c];
    float y = (scan_y[i] + value[i] * skip[c]) * siluf(g);
    ycb[i] = (bf16)y;
}

// ---------------- host launcher ----------------
static inline int cdiv(int a, int b) { return (a + b - 1) / b; }

extern "C" void kernel_launch(void* const* d_in, const int* in_sizes, int n_in,
                              void* d_out, int out_size, void* d_ws, size_t ws_size,
                              hipStream_t stream) {
    const float* x       = (const float*)d_in[0];
    const float* W_in    = (const float*)d_in[1];
    const float* conv_w  = (const float*)d_in[2];
    const float* conv_b  = (const float*)d_in[3];
    const float* W_bc    = (const float*)d_in[4];
    const float* W_out   = (const float*)d_in[5];
    const float* skip    = (const float*)d_in[6];
    const float* b_scale = (const float*)d_in[7];
    const float* c_scale = (const float*)d_in[8];
    const float* A_log   = (const float*)d_in[9];
    const float* dt_bias = (const float*)d_in[10];
    float* out = (float*)d_out;

    char* ws = (char*)d_ws;
    size_t off = 0;
    auto alloc = [&](size_t bytes) -> void* {
        void* p = ws + off;
        off = (off + bytes + 255) & ~(size_t)255;
        return p;
    };

    bf16*  xb     = (bf16*) alloc((size_t)T    * DM   * sizeof(bf16));
    bf16*  WinT   = (bf16*) alloc((size_t)Ppad * DM   * sizeof(bf16));  // [Ppad, DM]
    bf16*  WbcT   = (bf16*) alloc((size_t)BCW  * DI   * sizeof(bf16));  // [BCW, DI]
    bf16*  WoutT  = (bf16*) alloc((size_t)DM   * DI   * sizeof(bf16));  // [DM, DI]
    float* proj   = (float*)alloc((size_t)T    * Ppad * sizeof(float)); // stride Ppad
    float* value  = (float*)alloc((size_t)T    * DI   * sizeof(float));
    bf16*  valb   = (bf16*) alloc((size_t)T    * DI   * sizeof(bf16));
    float* bc     = (float*)alloc((size_t)T    * BCW  * sizeof(float));
    float* scan_y = (float*)alloc((size_t)T    * DI   * sizeof(float));
    bf16*  ycb    = (bf16*) alloc((size_t)T    * DI   * sizeof(bf16));
    (void)ws_size; (void)in_sizes; (void)n_in; (void)out_size;

    // 1) activations -> bf16; weights -> transposed bf16 [No, K]
    int n = T * DM;
    k_cvt_bf16<<<cdiv(n, 256), 256, 0, stream>>>(x, xb, n);
    {
        dim3 g(DM / 32, Ppad / 32);
        k_transpose_cvt<<<g, 256, 0, stream>>>(W_in, WinT, DM, P, Ppad);
    }
    {
        dim3 g(DI / 32, BCW / 32);
        k_transpose_cvt<<<g, 256, 0, stream>>>(W_bc, WbcT, DI, BCW, BCW);
    }
    {
        dim3 g(DI / 32, DM / 32);
        k_transpose_cvt<<<g, 256, 0, stream>>>(W_out, WoutT, DI, DM, DM);
    }

    // 2) proj = x @ W_in   (M=T, No=Ppad, K=DM)
    {
        int waves = (T / 64) * (Ppad / 32);
        k_gemm_bf16<<<cdiv(waves, 8), 256, 0, stream>>>(proj, xb, WinT, T, Ppad, DM);
    }

    // 3) causal conv + silu
    k_conv_silu<<<cdiv(T * DI, 256), 256, 0, stream>>>(proj, conv_w, conv_b, value, valb);

    // 4) bc = value @ W_bc   (M=T, No=BCW, K=DI)
    {
        int waves = (T / 64) * (BCW / 32);
        k_gemm_bf16<<<cdiv(waves, 8), 256, 0, stream>>>(bc, valb, WbcT, T, BCW, DI);
    }

    // 5) rms-norm + scales (in place on bc)
    k_rmsnorm<<<cdiv(T * H * 4, 8), 256, 0, stream>>>(bc, b_scale, c_scale);

    // 6) sequential scan, one block per head
    k_scan<<<H, 256, 0, stream>>>(proj, bc, value, A_log, dt_bias, scan_y);

    // 7) gate/skip combine -> bf16
    k_combine<<<cdiv(T * DI, 256), 256, 0, stream>>>(proj, scan_y, value, skip, ycb);

    // 8) out = yc @ W_out   (M=T, No=DM, K=DI)
    {
        int waves = (T / 64) * (DM / 32);
        k_gemm_bf16<<<cdiv(waves, 8), 256, 0, stream>>>(out, ycb, WoutT, T, DM, DI);
    }
}
